// PostProcess_90933047591168
// MI455X (gfx1250) — compile-verified, matrix operand order checked by reading, not run on hardware
//
#include <hip/hip_runtime.h>
#include <hip/hip_bf16.h>
#include <math.h>

#define NUM_QUERIES 10000
#define NUM_CLASSES 2048
#define IOU_THRESH 0.5f
#define NSORT 16384                 // pow2 padding for bitonic sort
#define NWORDS 313                  // ceil(10000/32) suppression words per row

typedef __attribute__((ext_vector_type(16))) _Float16 v16h;
typedef __attribute__((ext_vector_type(8)))  float    v8f;
typedef __attribute__((ext_vector_type(4)))  unsigned int u32x4;
typedef __attribute__((ext_vector_type(8)))  int      i32x8;
typedef __attribute__((ext_vector_type(4)))  int      i32x4;

// ---------------------------------------------------------------------------
// Kernel 1: per-row softmax stats via WMMA ones-GEMM.
// One wave (32 lanes) handles 16 rows. Lane L<16 covers row L, K in
// {0..7,16..23} (mod 32); lane L+16 covers the complementary K halves —
// exactly the 16-bit A-matrix 16x32 VGPR layout. B = ones => D row sums.
// score = exp(rowmax)/sum(exp(row)); label = argmax (first occurrence).
// ---------------------------------------------------------------------------
__global__ void softmax_stats_kernel(const float* __restrict__ logits,
                                     float* __restrict__ scores,
                                     float* __restrict__ labels) {
    const int lane = threadIdx.x & 31;
    const int wave = (blockIdx.x * blockDim.x + threadIdx.x) >> 5;
    const int rowBase = wave * 16;
    if (rowBase >= NUM_QUERIES) return;            // wave-uniform: EXEC stays all-1s

    const int r = rowBase + (lane & 15);
    const int kShift = (lane >> 4) * 8;            // 0 for lanes 0-15, 8 for 16-31
    const float* __restrict__ row = logits + (size_t)r * NUM_CLASSES;

    float maxv = -INFINITY;
    int   maxi = 0;
    v8f acc = {};
    v16h ones;
#pragma unroll
    for (int e = 0; e < 16; ++e) ones[e] = (_Float16)1.0f;

    for (int k0 = 0; k0 < NUM_CLASSES; k0 += 32) {
        const int ka = k0 + kShift;
        const int kb = ka + 16;
        float4 x0 = *reinterpret_cast<const float4*>(row + ka);
        float4 x1 = *reinterpret_cast<const float4*>(row + ka + 4);
        float4 x2 = *reinterpret_cast<const float4*>(row + kb);
        float4 x3 = *reinterpret_cast<const float4*>(row + kb + 4);
        const float v[16] = {x0.x,x0.y,x0.z,x0.w, x1.x,x1.y,x1.z,x1.w,
                             x2.x,x2.y,x2.z,x2.w, x3.x,x3.y,x3.z,x3.w};
        v16h a;
#pragma unroll
        for (int e = 0; e < 16; ++e) {
            const int kk = (e < 8) ? (ka + e) : (kb + (e - 8));
            const float x = v[e];
            if (x > maxv) { maxv = x; maxi = kk; }   // increasing kk => first-max kept
            a[e] = (_Float16)__expf(x);
        }
        // D = A(16x32 f16) * ones(32x16) + C : every column of D = row sum
        acc = __builtin_amdgcn_wmma_f32_16x16x32_f16(
            /*neg_a=*/false, a, /*neg_b=*/false, ones,
            /*c_mod=*/(short)0, acc, /*reuse_a=*/false, /*reuse_b=*/false);
    }

    // Merge the two K-halves of each row (partner lane = lane ^ 16).
    const float omax = __shfl_xor(maxv, 16, 32);
    const int   oidx = __shfl_xor(maxi, 16, 32);
    if (omax > maxv || (omax == maxv && oidx < maxi)) { maxv = omax; maxi = oidx; }

    if (lane < 16) labels[rowBase + lane] = (float)maxi;

    // C/D layout: acc[j] holds sum(row j) on lanes 0-15, sum(row j+8) on 16-31.
#pragma unroll
    for (int j = 0; j < 8; ++j) {
        const int src = (lane < 16) ? j : (8 + j);
        const float mj = __shfl(maxv, src, 32);
        const float s = __expf(mj) / acc[j];       // = 1/sum(exp(x - max))
        if (lane == 0)       scores[rowBase + j]     = s;
        else if (lane == 16) scores[rowBase + 8 + j] = s;
    }
}

// ---------------------------------------------------------------------------
// Kernel 2: box cxcywh -> xyxy, clip [0,1], scale by (w,h,w,h)
// ---------------------------------------------------------------------------
__global__ void box_kernel(const float* __restrict__ pb,
                           const long long* __restrict__ ts,   // int64 [h, w]
                           float* __restrict__ out) {
    const int i = blockIdx.x * blockDim.x + threadIdx.x;
    if (i >= NUM_QUERIES) return;
    const float4 b = reinterpret_cast<const float4*>(pb)[i];
    const float ih = (float)ts[0], iw = (float)ts[1];
    const float x0 = fminf(fmaxf(b.x - 0.5f * b.z, 0.f), 1.f) * iw;
    const float y0 = fminf(fmaxf(b.y - 0.5f * b.w, 0.f), 1.f) * ih;
    const float x1 = fminf(fmaxf(b.x + 0.5f * b.z, 0.f), 1.f) * iw;
    const float y1 = fminf(fmaxf(b.y + 0.5f * b.w, 0.f), 1.f) * ih;
    reinterpret_cast<float4*>(out)[i] = make_float4(x0, y0, x1, y1);
}

// ---------------------------------------------------------------------------
// Kernel 3: argmax over 2048 class logits
// ---------------------------------------------------------------------------
__global__ void cls_argmax_kernel(const float* __restrict__ cls, float* __restrict__ out) {
    __shared__ float smax[256];
    __shared__ int   sidx[256];
    const int t = threadIdx.x;
    float m = -INFINITY; int mi = 0;
    for (int i = t; i < NUM_CLASSES; i += 256) {
        const float v = cls[i];
        if (v > m) { m = v; mi = i; }
    }
    smax[t] = m; sidx[t] = mi;
    __syncthreads();
    for (int s = 128; s > 0; s >>= 1) {
        if (t < s) {
            const float om = smax[t + s]; const int oi = sidx[t + s];
            if (om > smax[t] || (om == smax[t] && oi < sidx[t])) { smax[t] = om; sidx[t] = oi; }
        }
        __syncthreads();
    }
    if (t == 0) out[0] = (float)sidx[0];
}

// ---------------------------------------------------------------------------
// Kernel 4: zero the keep mask region
// ---------------------------------------------------------------------------
__global__ void zero_keep_kernel(float* __restrict__ keep) {
    const int i = blockIdx.x * blockDim.x + threadIdx.x;
    if (i < NUM_QUERIES) keep[i] = 0.0f;
}

// ---------------------------------------------------------------------------
// Kernel 5: single-WG bitonic sort of (score, idx) desc in 128KB dynamic LDS
// (CDNA5 allows up to 320KB per WG), then gather sorted boxes + areas.
// Scores are staged into the LDS key region by a single Tensor-Data-Mover
// DMA (tensor_load_to_lds, 10000x1 tile of 4B elements) issued by wave 0.
// ---------------------------------------------------------------------------
__global__ void nms_sort_kernel(const float* __restrict__ scores,
                                const float* __restrict__ boxes,     // scaled xyxy
                                int* __restrict__ sortedIdx,
                                float4* __restrict__ sortedBoxes,
                                float* __restrict__ areas) {
    extern __shared__ float smem[];
    float* k  = smem;                 // NSORT keys
    int*   id = (int*)(smem + NSORT); // NSORT indices

#if defined(__has_builtin) && __has_builtin(__builtin_amdgcn_tensor_load_to_lds)
    if (threadIdx.x < 32) {                                  // wave 0 issues the DMA
        const unsigned long long gaddr = (unsigned long long)(uintptr_t)scores;
        // flat LDS-aperture address truncates to the LDS byte offset (ISA 10.2)
        const unsigned int ldsOff = (unsigned int)(uintptr_t)k;
        u32x4 g0;
        g0[0] = 1u;                                          // count=1, user descriptor
        g0[1] = ldsOff;                                      // lds_addr [63:32]
        g0[2] = (unsigned int)(gaddr & 0xffffffffull);       // global_addr[31:0]
        g0[3] = (unsigned int)((gaddr >> 32) & 0x01ffffffull)
              | (2u << 30);                                  // global_addr[56:32] | type=2
        i32x8 g1;
        g1[0] = (2 << 16);                                   // data_size = 4 bytes
        g1[1] = (int)((NUM_QUERIES & 0xffff) << 16);         // tensor_dim0[15:0]
        g1[2] = (1 << 16);                                   // tensor_dim1 = 1
        g1[3] = (int)((NUM_QUERIES & 0xffff) << 16);         // tile_dim0 = 10000
        g1[4] = 1;                                           // tile_dim1 = 1
        g1[5] = NUM_QUERIES;                                 // tensor_dim0_stride
        g1[6] = 0;
        g1[7] = 0;
        const i32x4 z4 = {0, 0, 0, 0};                       // <=2D: groups 2/3 unused
        const i32x8 z8 = {0, 0, 0, 0, 0, 0, 0, 0};           // 6-arg form: extra group
        __builtin_amdgcn_tensor_load_to_lds(g0, g1, z4, z4, z8, 0);
        __builtin_amdgcn_s_wait_tensorcnt(0);                // s_wait_tensorcnt 0
    }
    __syncthreads();
    for (int t = threadIdx.x; t < NSORT; t += blockDim.x) {
        if (t >= NUM_QUERIES) k[t] = -INFINITY;              // pad tail
        id[t] = t;
    }
#else
    for (int t = threadIdx.x; t < NSORT; t += blockDim.x) {
        k[t]  = (t < NUM_QUERIES) ? scores[t] : -INFINITY;
        id[t] = t;
    }
#endif
    __syncthreads();

    for (int size = 2; size <= NSORT; size <<= 1) {
        for (int stride = size >> 1; stride > 0; stride >>= 1) {
            for (int t = threadIdx.x; t < NSORT / 2; t += blockDim.x) {
                const int pos = 2 * t - (t & (stride - 1));
                const int a = pos, b = pos + stride;
                const bool descSeg = ((pos & size) == 0);   // final order: descending
                const float ka = k[a], kb = k[b];
                const int   ia = id[a], ib = id[b];
                const bool aBeforeB = (ka > kb) || (ka == kb && ia < ib); // stable
                const bool doSwap = descSeg ? (!aBeforeB) : aBeforeB;
                if (doSwap) { k[a] = kb; k[b] = ka; id[a] = ib; id[b] = ia; }
            }
            __syncthreads();
        }
    }

    for (int t = threadIdx.x; t < NUM_QUERIES; t += blockDim.x) {
        const int j = id[t];
        sortedIdx[t] = j;
        const float4 bx = reinterpret_cast<const float4*>(boxes)[j];
        sortedBoxes[t] = bx;
        areas[t] = fmaxf(bx.z - bx.x, 0.f) * fmaxf(bx.w - bx.y, 0.f);
    }
}

// ---------------------------------------------------------------------------
// Kernel 6: fully parallel suppression bit-matrix: supp[i][w] bit jj set iff
// IoU(sorted_i, sorted_{32w+jj}) > thresh and j > i.  10000x313 u32 = 12.5MB
// (resident in the 192MB L2).
// ---------------------------------------------------------------------------
__global__ void nms_iou_kernel(const float4* __restrict__ boxes,
                               const float* __restrict__ areas,
                               unsigned int* __restrict__ supp) {
    const long long gid = (long long)blockIdx.x * blockDim.x + threadIdx.x;
    if (gid >= (long long)NUM_QUERIES * NWORDS) return;
    const int i = (int)(gid / NWORDS);
    const int w = (int)(gid % NWORDS);
    const int jbase = w << 5;
    __builtin_prefetch(&boxes[jbase], 0, 0);       // global_prefetch_b8
    const float4 bi = boxes[i];
    const float  ai = areas[i];
    unsigned int bits = 0u;
#pragma unroll 4
    for (int jj = 0; jj < 32; ++jj) {
        const int j = jbase + jj;
        if (j < NUM_QUERIES && j > i) {
            const float4 bj = boxes[j];
            const float iw = fmaxf(fminf(bi.z, bj.z) - fmaxf(bi.x, bj.x), 0.f);
            const float ih = fmaxf(fminf(bi.w, bj.w) - fmaxf(bi.y, bj.y), 0.f);
            const float inter = iw * ih;
            const float iou = inter / (ai + areas[j] - inter);
            if (iou > IOU_THRESH) bits |= (1u << jj);
        }
    }
    supp[(long long)i * NWORDS + w] = bits;
}

// ---------------------------------------------------------------------------
// Kernel 7: single-wave greedy scan over the bit-matrix (wave32 lockstep).
// removed mask lives in LDS (313 words); alive boxes scatter keep=1 at their
// original index and OR their suppression row into the mask.
// ---------------------------------------------------------------------------
__global__ void nms_reduce_kernel(const unsigned int* __restrict__ supp,
                                  const int* __restrict__ sortedIdx,
                                  float* __restrict__ keep) {
    __shared__ unsigned int removed[NWORDS];
    const int lane = threadIdx.x;                  // 32 threads
    for (int w = lane; w < NWORDS; w += 32) removed[w] = 0u;
    __syncthreads();
    for (int i = 0; i < NUM_QUERIES; ++i) {
        const unsigned int word = removed[i >> 5]; // LDS broadcast, wave-uniform
        const bool alive = ((word >> (i & 31)) & 1u) == 0u;
        if (alive) {
            if (lane == 0) keep[sortedIdx[i]] = 1.0f;
            const unsigned int* row = supp + (long long)i * NWORDS;
            for (int w = lane; w < NWORDS; w += 32) removed[w] |= row[w];
        }
        __syncthreads();
    }
}

// ---------------------------------------------------------------------------
extern "C" void kernel_launch(void* const* d_in, const int* in_sizes, int n_in,
                              void* d_out, int out_size, void* d_ws, size_t ws_size,
                              hipStream_t stream) {
    const float*     logits = (const float*)d_in[0];       // (1,10000,2048) f32
    const float*     pboxes = (const float*)d_in[1];       // (1,10000,4)    f32
    const float*     clslog = (const float*)d_in[2];       // (1,2048)       f32
    const long long* tsz    = (const long long*)d_in[3];   // (1,2)          i64

    float* out    = (float*)d_out;                 // concatenated outputs:
    float* scores = out;                           // [0,      10000)
    float* labels = out + NUM_QUERIES;             // [10000,  20000)
    float* boxes  = out + 2 * NUM_QUERIES;         // [20000,  60000)
    float* keep   = out + 6 * NUM_QUERIES;         // [60000,  70000)
    float* clsp   = out + 7 * NUM_QUERIES;         // [70000,  70001)

    // workspace layout
    char* ws = (char*)d_ws;
    int*          sortedIdx   = (int*)ws;                                  //  64 KB
    float4*       sortedBoxes = (float4*)(ws + 65536);                     // 256 KB
    float*        areas       = (float*)(ws + 65536 + 262144);             //  64 KB
    unsigned int* supp        = (unsigned int*)(ws + 65536 + 262144 + 65536); // 12.52 MB

    // 1) zero keep mask (harness poisons d_out)
    zero_keep_kernel<<<(NUM_QUERIES + 255) / 256, 256, 0, stream>>>(keep);

    // 2) softmax stats: 625 waves of 16 rows, 8 waves/block -> 79 blocks
    softmax_stats_kernel<<<79, 256, 0, stream>>>(logits, scores, labels);

    // 3) box transform
    box_kernel<<<(NUM_QUERIES + 255) / 256, 256, 0, stream>>>(pboxes, tsz, boxes);

    // 4) class argmax
    cls_argmax_kernel<<<1, 256, 0, stream>>>(clslog, clsp);

    // 5) sort by score desc (single WG, 128 KB dynamic LDS, TDM-staged keys)
    nms_sort_kernel<<<1, 1024, NSORT * 8, stream>>>(scores, boxes,
                                                    sortedIdx, sortedBoxes, areas);

    // 6) parallel suppression bit-matrix
    const long long iouThreads = (long long)NUM_QUERIES * NWORDS;
    nms_iou_kernel<<<(unsigned)((iouThreads + 255) / 256), 256, 0, stream>>>(
        sortedBoxes, areas, supp);

    // 7) single-wave greedy reduction -> keep
    nms_reduce_kernel<<<1, 32, 0, stream>>>(supp, sortedIdx, keep);
}
